// MSAClass_12025908429170
// MI455X (gfx1250) — matrix operand order repaired and checked by reading, not compiled, not run on hardware
//
#include <hip/hip_runtime.h>

#define B_  8
#define S_  1024
#define D_  1024
#define H_  16
#define HD_ 64

typedef __attribute__((ext_vector_type(16))) __bf16       v16bf;
typedef __attribute__((ext_vector_type(8)))  float        v8f;
typedef __attribute__((ext_vector_type(4)))  float        v4f;
typedef __attribute__((ext_vector_type(4)))  unsigned int v4u;
typedef __attribute__((ext_vector_type(8)))  unsigned int v8u;

union BF16x16 { v16bf v; v4u u[2]; };

__device__ __forceinline__ unsigned short f2bf(float f) {
  unsigned u = __float_as_uint(f);
  unsigned r = u + 0x7FFFu + ((u >> 16) & 1u);   // round-to-nearest-even
  return (unsigned short)(r >> 16);
}

// DPP16 xor-butterfly exchange within each 16-lane row: ctrl = 0x160 | mask.
template <int CTRL>
__device__ __forceinline__ float dppf(float x) {
  return __int_as_float(
      __builtin_amdgcn_update_dpp(0, __float_as_int(x), CTRL, 0xf, 0xf, true));
}
__device__ __forceinline__ float rowmax16(float m) {
  m = fmaxf(m, dppf<0x161>(m));   // row_xmask:1
  m = fmaxf(m, dppf<0x162>(m));   // row_xmask:2
  m = fmaxf(m, dppf<0x164>(m));   // row_xmask:4
  m = fmaxf(m, dppf<0x168>(m));   // row_xmask:8
  return m;
}

// ---------------------------------------------------------------------------
// TDM: issue a 2-D bf16 tile DMA (global -> LDS) via a Tensor DMA Descriptor.
// Bit layout per CDNA5 ISA §8.3/§8.4 (group0 128b, group1 256b), 2-group form
// (VADDR2/3 = NULL -> tensors up to 2D). Tracked by TENSORcnt, completes
// in-order with other tensor ops of the issuing wave.
// ---------------------------------------------------------------------------
__device__ __forceinline__ void tdm_load_2d_bf16(
    unsigned lds_off, const void* gaddr,
    unsigned tensor_d0, unsigned tensor_d1,
    unsigned tile_d0, unsigned tile_d1, unsigned stride_d0)
{
  const unsigned long long ga = (unsigned long long)(uintptr_t)gaddr;
  v4u g0;
  g0[0] = 1u;                                         // count=1, user mode
  g0[1] = lds_off;                                    // lds_addr (bytes)
  g0[2] = (unsigned)(ga & 0xFFFFFFFFu);               // global_addr[31:0]
  g0[3] = (unsigned)((ga >> 32) & 0x01FFFFFFu)        // global_addr[56:32]
          | 0x80000000u;                              // type=2 ("image")
  v8u g1;
  g1[0] = 0x00010000u;                                // data_size=1 (2 bytes)
  g1[1] = (tensor_d0 & 0xFFFFu) << 16;                // tensor_dim0[15:0]
  g1[2] = (tensor_d0 >> 16) | ((tensor_d1 & 0xFFFFu) << 16);
  g1[3] = (tensor_d1 >> 16) | (tile_d0 << 16);        // tile_dim0
  g1[4] = tile_d1;                                    // tile_dim1 (tile_dim2=0)
  g1[5] = stride_d0;                                  // tensor_dim0_stride lo32
  g1[6] = 0;
  g1[7] = 0;
  asm volatile("tensor_load_to_lds %0, %1" :: "s"(g0), "s"(g1) : "memory");
}

// ---------------------------------------------------------------------------
// Kernel 0: fp32 -> bf16 convert, 8 elements/thread (b128 in/out).
// ---------------------------------------------------------------------------
__global__ __launch_bounds__(256) void bf16_cvt8_kernel(
    const float* __restrict__ src, unsigned short* __restrict__ dst, int n8) {
  int i = blockIdx.x * 256 + threadIdx.x;
  if (i < n8) {
    const v4f* s = (const v4f*)src + (size_t)i * 2;
    v4f a = s[0], b = s[1];
    union { v4u u; unsigned short h[8]; } pk;
    pk.h[0] = f2bf(a.x); pk.h[1] = f2bf(a.y);
    pk.h[2] = f2bf(a.z); pk.h[3] = f2bf(a.w);
    pk.h[4] = f2bf(b.x); pk.h[5] = f2bf(b.y);
    pk.h[6] = f2bf(b.z); pk.h[7] = f2bf(b.w);
    ((v4u*)dst)[i] = pk.u;
  }
}

// ---------------------------------------------------------------------------
// Kernel 1: QKV projection on the WMMA pipe. One wave32 per 16-query tile.
//   y = x @ W^T + b per head.  A = x tile (16x64 bf16), B = W^T chunks.
//   q,k written [B,H,S,HD]; v written transposed [B,H,HD,S] (b128 packed).
// ---------------------------------------------------------------------------
__global__ __launch_bounds__(256) void qkv_wmma_kernel(
    const unsigned short* __restrict__ xb,
    const unsigned short* __restrict__ wqb,
    const unsigned short* __restrict__ wkb,
    const unsigned short* __restrict__ wvb,
    const float* __restrict__ bq, const float* __restrict__ bk,
    const float* __restrict__ bv,
    unsigned short* __restrict__ q, unsigned short* __restrict__ k,
    unsigned short* __restrict__ vt)
{
  const int lane = threadIdx.x & 31;
  const int widx = threadIdx.x >> 5;
  const int hi   = lane >> 4;
  const int ln   = lane & 15;
  const int w    = blockIdx.x * 8 + widx;       // B*H*(S/16) = 8192 waves
  const int qt   = w & 63;
  const int h    = (w >> 6) & (H_ - 1);
  const int b    = w >> 10;
  const int s0   = qt * 16;
  const size_t bh = (size_t)b * H_ + h;

  BF16x16 xa[2];
#pragma unroll
  for (int c = 0; c < 2; ++c) {
    const unsigned short* p =
        xb + ((size_t)b * S_ + s0 + ln) * D_ + h * HD_ + c * 32 + hi * 16;
    xa[c].u[0] = *(const v4u*)(p);
    xa[c].u[1] = *(const v4u*)(p + 8);
  }

  const unsigned short* wmat[3] = { wqb + (size_t)h * HD_ * HD_,
                                    wkb + (size_t)h * HD_ * HD_,
                                    wvb + (size_t)h * HD_ * HD_ };
  const float* bias[3] = { bq + h * HD_, bk + h * HD_, bv + h * HD_ };
  unsigned short* qk[2] = { q + (bh * S_ + s0) * HD_,
                            k + (bh * S_ + s0) * HD_ };

#pragma unroll
  for (int pidx = 0; pidx < 3; ++pidx) {
#pragma unroll
    for (int t = 0; t < 4; ++t) {
      const int e = t * 16 + ln;
      v8f c = {};
#pragma unroll
      for (int c2 = 0; c2 < 2; ++c2) {
        BF16x16 wb;
        const unsigned short* p = wmat[pidx] + (size_t)e * HD_ + c2 * 32 + hi * 16;
        wb.u[0] = *(const v4u*)(p);
        wb.u[1] = *(const v4u*)(p + 8);
        c = __builtin_amdgcn_wmma_f32_16x16x32_bf16(false, xa[c2].v, false, wb.v,
                                                    (short)0, c, false, false);
      }
      const float be = bias[pidx][e];
      if (pidx < 2) {
        unsigned short* dst = qk[pidx];
#pragma unroll
        for (int i = 0; i < 8; ++i)
          dst[(size_t)(hi * 8 + i) * HD_ + e] = f2bf(c[i] + be);
      } else {
        union { v4u u; unsigned short s[8]; } pk;
#pragma unroll
        for (int i = 0; i < 8; ++i) pk.s[i] = f2bf(c[i] + be);
        *(v4u*)(vt + (bh * HD_ + e) * S_ + s0 + hi * 8) = pk.u;
      }
    }
  }
}

// ---------------------------------------------------------------------------
// Kernel 2: flash attention, double-buffered TDM feed. 8 waves/block share
//   (b,h): wave 0 issues tensor_load_to_lds for the NEXT 32-key K/V tiles
//   into the alternate LDS buffer, then s_wait_tensorcnt(2) (in-order ->
//   previous pair has landed), so DMA overlaps a full iteration of WMMA +
//   softmax. Zero per-wave VMEM in the hot loop; row-max via DPP xor
//   butterflies; row-sum via one extra WMMA (P @ ones).
// ---------------------------------------------------------------------------
__global__ __launch_bounds__(256) void flash_attn_kernel(
    const unsigned short* __restrict__ qb,
    const unsigned short* __restrict__ kb,
    const unsigned short* __restrict__ vtb,
    float* __restrict__ out)
{
  const int lane = threadIdx.x & 31;
  const int widx = threadIdx.x >> 5;
  const int hi   = lane >> 4;
  const int ln   = lane & 15;
  const int w    = blockIdx.x * 8 + widx;
  const int qt   = w & 63;
  const int h    = (w >> 6) & (H_ - 1);
  const int b    = w >> 10;
  const int s0   = qt * 16;
  const size_t bh = (size_t)b * H_ + h;

  __shared__ __align__(16) unsigned short ldsK[2][32 * HD_];   // 2 x 4KB
  __shared__ __align__(16) unsigned short ldsV[2][HD_ * 32];   // 2 x 4KB
  __shared__ __align__(16) unsigned short pstage[8][16 * 32];  // 8KB
  unsigned short* pst = pstage[widx];

  const unsigned short* qbase = qb  + (bh * S_ + s0) * HD_;
  const unsigned short* kbase = kb  + bh * S_ * HD_;
  const unsigned short* vbase = vtb + bh * HD_ * S_;

  // Flat LDS aperture: addr[31:0] is the LDS byte offset -> D#.lds_addr.
  const unsigned kOff[2] = { (unsigned)(uintptr_t)(void*)ldsK[0],
                             (unsigned)(uintptr_t)(void*)ldsK[1] };
  const unsigned vOff[2] = { (unsigned)(uintptr_t)(void*)ldsV[0],
                             (unsigned)(uintptr_t)(void*)ldsV[1] };

  // Q tile as two bf16 A-matrices (K-chunks of 32).
  BF16x16 qa[2];
#pragma unroll
  for (int c = 0; c < 2; ++c) {
    const unsigned short* p = qbase + ln * HD_ + c * 32 + hi * 16;
    qa[c].u[0] = *(const v4u*)(p);
    qa[c].u[1] = *(const v4u*)(p + 8);
  }

  BF16x16 onesb;                                  // bf16 1.0 splat
  onesb.u[0] = (v4u){0x3F803F80u, 0x3F803F80u, 0x3F803F80u, 0x3F803F80u};
  onesb.u[1] = onesb.u[0];

  v8f acc[4] = { v8f{}, v8f{}, v8f{}, v8f{} };
  v8f lv = {};
  float mrow[8];
#pragma unroll
  for (int i = 0; i < 8; ++i) mrow[i] = -3.0e38f;

  const float SCALE = 0.125f;                     // 1/sqrt(64)

  // ---- Prologue: DMA the first K/V tiles into buffer 0.
  if (widx == 0) {
    tdm_load_2d_bf16(kOff[0], kbase, HD_, 32, HD_, 32, HD_);
    tdm_load_2d_bf16(vOff[0], vbase, 32, HD_, 32, HD_, S_);
  }

  int it = 0;
  for (int k0 = 0; k0 < S_; k0 += 32, it ^= 1) {
    if (widx == 0) {
      if (k0 + 32 < S_) {
        // Issue next tiles first, then wait only for the previous pair:
        // TENSORcnt is in-order, so cnt<=2 => the two older ops completed.
        tdm_load_2d_bf16(kOff[it ^ 1], kbase + (size_t)(k0 + 32) * HD_,
                         HD_, 32, HD_, 32, HD_);
        tdm_load_2d_bf16(vOff[it ^ 1], vbase + k0 + 32, 32, HD_, 32, HD_, S_);
        __builtin_amdgcn_s_wait_tensorcnt(2);
      } else {
        __builtin_amdgcn_s_wait_tensorcnt(0);
      }
    }
    __syncthreads();                              // buf[it] ready for all waves

    const unsigned short* ldsKc = ldsK[it];
    const unsigned short* ldsVc = ldsV[it];

    // ---- S tiles from LDS: s1 = Q @ K[k0..+15]^T, s2 = Q @ K[k0+16..+31]^T.
    v8f s1 = {}, s2 = {};
#pragma unroll
    for (int c = 0; c < 2; ++c) {
      BF16x16 kt1, kt2;
      const unsigned short* p1 = ldsKc + ln * HD_ + c * 32 + hi * 16;
      kt1.u[0] = *(const v4u*)(p1);
      kt1.u[1] = *(const v4u*)(p1 + 8);
      const unsigned short* p2 = p1 + 16 * HD_;
      kt2.u[0] = *(const v4u*)(p2);
      kt2.u[1] = *(const v4u*)(p2 + 8);
      s1 = __builtin_amdgcn_wmma_f32_16x16x32_bf16(false, qa[c].v, false, kt1.v,
                                                   (short)0, s1, false, false);
      s2 = __builtin_amdgcn_wmma_f32_16x16x32_bf16(false, qa[c].v, false, kt2.v,
                                                   (short)0, s2, false, false);
    }

    // ---- Online softmax. Row i+hi*8 lives in VGPR i, column = lane&15.
#pragma unroll
    for (int i = 0; i < 8; ++i) {
      float e1 = s1[i] * SCALE;
      float e2 = s2[i] * SCALE;
      float m  = rowmax16(fmaxf(e1, e2));
      float mn   = fmaxf(mrow[i], m);
      float corr = __expf(mrow[i] - mn);
      mrow[i] = mn;
      const int prow = hi * 8 + i;
      pst[prow * 32 + ln]      = f2bf(__expf(e1 - mn));
      pst[prow * 32 + 16 + ln] = f2bf(__expf(e2 - mn));
      acc[0][i] *= corr; acc[1][i] *= corr;
      acc[2][i] *= corr; acc[3][i] *= corr;
      lv[i]     *= corr;
    }

    asm volatile("" ::: "memory");                // LDS in-order within wave

    // ---- Reload P in the bf16 A-layout (16x32).
    BF16x16 pa;
    {
      const unsigned short* lp = pst + ln * 32 + hi * 16;
      pa.u[0] = *(const v4u*)(lp);
      pa.u[1] = *(const v4u*)(lp + 8);
    }

    // ---- Row sums: lv += P @ ones (one WMMA, no lane shuffles).
    lv = __builtin_amdgcn_wmma_f32_16x16x32_bf16(false, pa.v, false, onesb.v,
                                                 (short)0, lv, false, false);

    // ---- O += P @ V from the shared LDS V^T tile.
#pragma unroll
    for (int t = 0; t < 4; ++t) {
      BF16x16 vbm;
      const unsigned short* vp = ldsVc + (t * 16 + ln) * 32 + hi * 16;
      vbm.u[0] = *(const v4u*)(vp);
      vbm.u[1] = *(const v4u*)(vp + 8);
      acc[t] = __builtin_amdgcn_wmma_f32_16x16x32_bf16(false, pa.v, false, vbm.v,
                                                       (short)0, acc[t], false, false);
    }

    __syncthreads();   // all reads of buf[it] done before it is re-filled
  }

  // ---- Normalize and write out [B,S,D] fp32.
  float* ob = out + ((size_t)b * S_ + s0) * D_ + h * HD_;
#pragma unroll
  for (int t = 0; t < 4; ++t) {
#pragma unroll
    for (int i = 0; i < 8; ++i) {
      const int row = hi * 8 + i;
      ob[(size_t)row * D_ + t * 16 + ln] = acc[t][i] / lv[i];
    }
  }
}

// ---------------------------------------------------------------------------
extern "C" void kernel_launch(void* const* d_in, const int* in_sizes, int n_in,
                              void* d_out, int out_size, void* d_ws, size_t ws_size,
                              hipStream_t stream) {
  (void)in_sizes; (void)n_in; (void)out_size; (void)ws_size;
  const float* seq = (const float*)d_in[0];
  const float* Wq  = (const float*)d_in[1];
  const float* bq  = (const float*)d_in[2];
  const float* Wk  = (const float*)d_in[3];
  const float* bk  = (const float*)d_in[4];
  const float* Wv  = (const float*)d_in[5];
  const float* bv  = (const float*)d_in[6];

  const size_t n  = (size_t)B_ * S_ * D_;       // 8,388,608
  const size_t wn = (size_t)H_ * HD_ * HD_;     // 65,536
  unsigned short* xb  = (unsigned short*)d_ws;  // bf16 x    [B,S,D]
  unsigned short* qbf = xb  + n;                // bf16 Q    [B,H,S,HD]
  unsigned short* kbf = qbf + n;                // bf16 K    [B,H,S,HD]
  unsigned short* vtb = kbf + n;                // bf16 V^T  [B,H,HD,S]
  unsigned short* wqb = vtb + n;                // bf16 Wq   [H,HD,HD]
  unsigned short* wkb = wqb + wn;
  unsigned short* wvb = wkb + wn;

  bf16_cvt8_kernel<<<(int)(n  / 8 / 256), 256, 0, stream>>>(seq, xb,  (int)(n  / 8));
  bf16_cvt8_kernel<<<(int)(wn / 8 / 256), 256, 0, stream>>>(Wq,  wqb, (int)(wn / 8));
  bf16_cvt8_kernel<<<(int)(wn / 8 / 256), 256, 0, stream>>>(Wk,  wkb, (int)(wn / 8));
  bf16_cvt8_kernel<<<(int)(wn / 8 / 256), 256, 0, stream>>>(Wv,  wvb, (int)(wn / 8));

  const int waves = B_ * H_ * (S_ / 16);        // 8192 tiles
  qkv_wmma_kernel<<<waves / 8, 256, 0, stream>>>(
      xb, wqb, wkb, wvb, bq, bk, bv, qbf, kbf, vtb);
  flash_attn_kernel<<<waves / 8, 256, 0, stream>>>(qbf, kbf, vtb, (float*)d_out);
}
